// MoEHead_31499290149025
// MI455X (gfx1250) — matrix operand order repaired and checked by reading, not compiled, not run on hardware
//
#include <hip/hip_runtime.h>
#include <hip/hip_bf16.h>

// ---------------- Problem constants (match reference) ----------------
#define BTOK   65536
#define IN_DIM 1280
#define NEXP   8
#define H1     256
#define H2     128

// Tiling
#define MT     64          // tokens per workgroup
#define KC     32          // K chunk per WMMA staging step
#define XP     1288        // x_lds row pitch (bf16), padded (stride mod 64 banks != 0)
#define H1P    264         // h1_lds row pitch (bf16), padded
#define H2P    132         // h2_lds row pitch (f32), padded
#define WPP    20          // w_pk row pitch (u32): 16B-aligned AND bank-conflict-free
                           // (20*4 % 16 == 0; stride 20 mod 64 -> 16 distinct banks)

typedef __attribute__((ext_vector_type(8)))  float  v8f;
typedef __attribute__((ext_vector_type(16))) __bf16 v16bf;

union ABFrag { int4 i[2]; v16bf v; };

__device__ __forceinline__ unsigned f2bfbits(float f) {
    union { float f; unsigned u; } c; c.f = f;
    return ((c.u + 0x7FFFu + ((c.u >> 16) & 1u)) >> 16) & 0xFFFFu;   // RNE
}

#if __has_builtin(__builtin_amdgcn_cvt_pk_bf16_f32)
typedef __attribute__((ext_vector_type(2))) __bf16 v2bf;
__device__ __forceinline__ unsigned pack2(float lo, float hi) {
    union { v2bf v; unsigned u; } c;
    c.v = __builtin_amdgcn_cvt_pk_bf16_f32(lo, hi);   // v_cvt_pk_bf16_f32
    return c.u;
}
#else
__device__ __forceinline__ unsigned pack2(float lo, float hi) {
    return f2bfbits(lo) | (f2bfbits(hi) << 16);
}
#endif

__device__ __forceinline__ unsigned short f2bf(float f) {
    return (unsigned short)f2bfbits(f);
}
__device__ __forceinline__ float bf2f(unsigned short h) {
    union { unsigned u; float f; } c; c.u = ((unsigned)h) << 16;
    return c.f;
}

__global__ __launch_bounds__(256, 1)
void moe_head_fused(const float* __restrict__ x,
                    const float* __restrict__ W1, const float* __restrict__ b1,
                    const float* __restrict__ W2, const float* __restrict__ b2,
                    const float* __restrict__ W3, const float* __restrict__ b3,
                    const float* __restrict__ Wg, const float* __restrict__ bg,
                    float* __restrict__ out, int ntok)
{
    // ---------------- LDS (~255 KB of the 320 KB WGP budget) ----------------
    __shared__ __align__(16) unsigned short x_lds[MT * XP];     // 161.0 KB bf16 x tile
    __shared__ __align__(16) unsigned short h1_lds[MT * H1P];   //  33.0 KB bf16 h1
    __shared__ __align__(16) float          h2_lds[MT * H2P];   //  33.0 KB f32 h2
    __shared__ __align__(16) unsigned int   w_pk[H1 * WPP];     //  20.0 KB packed-B chunk
    __shared__ float b1_lds[H1];
    __shared__ float b2_lds[H2];
    __shared__ float w3_lds[H2];
    __shared__ float eo_lds[MT * NEXP];                         // expert outputs
    __shared__ float lg_lds[MT * NEXP];                         // gate logits
    __shared__ float b3_s;

    const int tid   = threadIdx.x;
    const int lane  = tid & 31;
    const int wave  = tid >> 5;
    const int l16   = lane & 15;
    const int khalf = lane >> 4;           // 0 or 1

    // layer-1 wave mapping: 4 M-subtiles x 2 N-halves
    const int m_base  = (wave & 3) * 16;
    const int n_base  = (wave >> 2) * 128; // 8 N-tiles of 16
    // layer-2 wave mapping: 4 M-subtiles x 2 N-halves of 64
    const int n_base2 = (wave >> 2) * 64;  // 4 N-tiles of 16

    const int wg = blockIdx.x;
    const float* xtile = x + (size_t)wg * MT * IN_DIM;

    // ---------------- Stage x tile -> bf16 LDS ----------------
    for (int i = tid; i < MT * (IN_DIM / 4); i += 256) {
        int row = i / (IN_DIM / 4);
        int c4  = (i % (IN_DIM / 4)) * 4;
        float4 v = *(const float4*)(xtile + row * IN_DIM + c4);
        uint2 s; s.x = pack2(v.x, v.y); s.y = pack2(v.z, v.w);
        *(uint2*)&x_lds[row * XP + c4] = s;
    }
    __syncthreads();

    // ---------------- Gate logits (reads x tile from LDS) ----------------
    for (int j = tid; j < MT * NEXP; j += 256) {
        int t = j >> 3, e = j & 7;
        float acc = bg[e];
        const unsigned short* xr = &x_lds[t * XP];
        for (int i = 0; i < IN_DIM; ++i)
            acc += bf2f(xr[i]) * Wg[i * NEXP + e];
        lg_lds[t * NEXP + e] = acc;
    }

    // ---------------- Expert loop ----------------
    for (int e = 0; e < NEXP; ++e) {
        __syncthreads();   // protect w3/b* LDS from previous iteration's readers

        // stage per-expert biases / W3
        b1_lds[tid] = b1[e * H1 + tid];
        if (tid < H2) { b2_lds[tid] = b2[e * H2 + tid]; w3_lds[tid] = W3[e * H2 + tid]; }
        if (tid == 0) b3_s = b3[e];

        // ======== Layer 1: [64,1280] x [1280,256] -> h1 ========
        v8f z; for (int r = 0; r < 8; ++r) z[r] = 0.0f;
        v8f acc1[8]; for (int n = 0; n < 8; ++n) acc1[n] = z;

        const float* W1e = W1 + (size_t)e * IN_DIM * H1;
        for (int k0 = 0; k0 < IN_DIM; k0 += KC) {
            // cooperative stage of W1 chunk [KC,256] fp32 -> packed bf16 B layout
            for (int s = 0; s < 4; ++s) {
                int item = tid + s * 256;          // 0..1023
                int kp   = item >> 6;              // k-pair 0..15
                int n    = (item & 63) * 4;        // 0..252
                const float* p = W1e + (size_t)(k0 + 2 * kp) * H1 + n;
                float4 ra = *(const float4*)p;
                float4 rb = *(const float4*)(p + H1);
                w_pk[(n + 0) * WPP + kp] = pack2(ra.x, rb.x);
                w_pk[(n + 1) * WPP + kp] = pack2(ra.y, rb.y);
                w_pk[(n + 2) * WPP + kp] = pack2(ra.z, rb.z);
                w_pk[(n + 3) * WPP + kp] = pack2(ra.w, rb.w);
            }
            if (k0 + KC < IN_DIM)  // prefetch next chunk (global_prefetch_b8)
                __builtin_prefetch((const char*)(W1e + (size_t)(k0 + KC) * H1) + tid * 128, 0, 1);
            __syncthreads();

            // A fragment: 16x32 bf16 from x_lds (documented 16-bit A layout)
            const unsigned short* ap = &x_lds[(m_base + l16) * XP + k0 + khalf * 8];
            ABFrag a; a.i[0] = *(const int4*)ap; a.i[1] = *(const int4*)(ap + 16);

            // manual unroll: reuse-A hint must be a literal constant
#define L1_STEP(NT, RA)                                                          \
            {                                                                    \
                const unsigned int* bp =                                         \
                    &w_pk[(n_base + (NT) * 16 + l16) * WPP + khalf * 4];         \
                ABFrag b; b.i[0] = *(const int4*)bp; b.i[1] = *(const int4*)(bp + 8); \
                acc1[NT] = __builtin_amdgcn_wmma_f32_16x16x32_bf16(              \
                    false, a.v, false, b.v, (short)0, acc1[NT], RA, false);      \
            }
            L1_STEP(0, false)
            L1_STEP(1, true)
            L1_STEP(2, true)
            L1_STEP(3, true)
            L1_STEP(4, true)
            L1_STEP(5, true)
            L1_STEP(6, true)
            L1_STEP(7, false)
#undef L1_STEP
            __syncthreads();
        }

        // bias + relu + write h1 as bf16
        for (int nt = 0; nt < 8; ++nt) {
            int n = n_base + nt * 16 + l16;
            float bias = b1_lds[n];
            for (int r = 0; r < 8; ++r) {
                int m = m_base + r + khalf * 8;
                float v = acc1[nt][r] + bias;
                h1_lds[m * H1P + n] = f2bf(v > 0.0f ? v : 0.0f);
            }
        }
        __syncthreads();

        // ======== Layer 2: [64,256] x [256,128] -> h2 ========
        v8f acc2[4]; for (int n = 0; n < 4; ++n) acc2[n] = z;
        const float* W2e = W2 + (size_t)e * H1 * H2;
        for (int k0 = 0; k0 < H1; k0 += KC) {
            for (int s = 0; s < 2; ++s) {
                int item = tid + s * 256;          // 0..511
                int kp   = item >> 5;              // 0..15
                int n    = (item & 31) * 4;        // 0..124
                const float* p = W2e + (size_t)(k0 + 2 * kp) * H2 + n;
                float4 ra = *(const float4*)p;
                float4 rb = *(const float4*)(p + H2);
                w_pk[(n + 0) * WPP + kp] = pack2(ra.x, rb.x);
                w_pk[(n + 1) * WPP + kp] = pack2(ra.y, rb.y);
                w_pk[(n + 2) * WPP + kp] = pack2(ra.z, rb.z);
                w_pk[(n + 3) * WPP + kp] = pack2(ra.w, rb.w);
            }
            __syncthreads();

            const unsigned short* ap = &h1_lds[(m_base + l16) * H1P + k0 + khalf * 8];
            ABFrag a; a.i[0] = *(const int4*)ap; a.i[1] = *(const int4*)(ap + 16);

#define L2_STEP(NT, RA)                                                          \
            {                                                                    \
                const unsigned int* bp =                                         \
                    &w_pk[(n_base2 + (NT) * 16 + l16) * WPP + khalf * 4];        \
                ABFrag b; b.i[0] = *(const int4*)bp; b.i[1] = *(const int4*)(bp + 8); \
                acc2[NT] = __builtin_amdgcn_wmma_f32_16x16x32_bf16(              \
                    false, a.v, false, b.v, (short)0, acc2[NT], RA, false);      \
            }
            L2_STEP(0, false)
            L2_STEP(1, true)
            L2_STEP(2, true)
            L2_STEP(3, false)
#undef L2_STEP
            __syncthreads();
        }

        // bias + relu + write h2 as f32
        for (int nt = 0; nt < 4; ++nt) {
            int n = n_base2 + nt * 16 + l16;
            float bias = b2_lds[n];
            for (int r = 0; r < 8; ++r) {
                int m = m_base + r + khalf * 8;
                float v = acc2[nt][r] + bias;
                h2_lds[m * H2P + n] = (v > 0.0f ? v : 0.0f);
            }
        }
        __syncthreads();

        // ======== Layer 3: h2 . W3[e] + b3[e] ========
        if (tid < MT) {
            float acc = b3_s;
            const float* hr = &h2_lds[tid * H2P];
            for (int n = 0; n < H2; ++n) acc += hr[n] * w3_lds[n];
            eo_lds[tid * NEXP + e] = acc;
        }
    }
    __syncthreads();

    // ---------------- Softmax gate + weighted combine ----------------
    if (tid < MT) {
        int token = wg * MT + tid;
        float l[NEXP];
        float mx = -3.4e38f;
        for (int e = 0; e < NEXP; ++e) { l[e] = lg_lds[tid * NEXP + e]; mx = l[e] > mx ? l[e] : mx; }
        float s = 0.0f;
        for (int e = 0; e < NEXP; ++e) { l[e] = __expf(l[e] - mx); s += l[e]; }
        float inv = 1.0f / s;
        float pred = 0.0f;
        for (int e = 0; e < NEXP; ++e) {
            float g = l[e] * inv;
            out[ntok + token * NEXP + e] = g;           // gate output
            pred += eo_lds[tid * NEXP + e] * g;
        }
        out[token] = pred;                               // predictions
    }
}

extern "C" void kernel_launch(void* const* d_in, const int* in_sizes, int n_in,
                              void* d_out, int out_size, void* d_ws, size_t ws_size,
                              hipStream_t stream) {
    const float* x  = (const float*)d_in[0];
    const float* W1 = (const float*)d_in[1];
    const float* b1 = (const float*)d_in[2];
    const float* W2 = (const float*)d_in[3];
    const float* b2 = (const float*)d_in[4];
    const float* W3 = (const float*)d_in[5];
    const float* b3 = (const float*)d_in[6];
    const float* Wg = (const float*)d_in[7];
    const float* bg = (const float*)d_in[8];
    float* out = (float*)d_out;

    int ntok = in_sizes[0] / IN_DIM;     // 65536
    int nblocks = ntok / MT;             // 1024

    moe_head_fused<<<nblocks, 256, 0, stream>>>(x, W1, b1, W2, b2, W3, b3, Wg, bg, out, ntok);
}
